// Mamba_TransformerEncoder_9165460210045
// MI455X (gfx1250) — compile-verified, matrix operand order checked by reading, not run on hardware
//
#include <hip/hip_runtime.h>
#include <stdint.h>

// ---------------- model constants ----------------
enum { BB = 4, LL = 512, DD = 512, NH = 8, FF = 2048, NL = 4 };
enum { DS = 64, DC = 4, DI = 1024, DTR = 32, HD = DD / NH };
enum { ROWS = BB * LL };           // 2048 token rows

// ---------------- vector types -------------------
typedef __attribute__((ext_vector_type(16))) __bf16    v16bf;
typedef __attribute__((ext_vector_type(8)))  float     v8f;
typedef __attribute__((ext_vector_type(8)))  uint16_t  u16x8;

union AFrag { v16bf bf; u16x8 h[2]; };

__device__ __forceinline__ uint16_t f2bf(float f) {
    union { float f; uint32_t u; } c; c.f = f;
    uint32_t u = c.u;
    uint32_t r = u + 0x7fffu + ((u >> 16) & 1u);
    return (uint16_t)(r >> 16);
}
__device__ __forceinline__ float bf2f(uint16_t h) {
    union { uint32_t u; float f; } c; c.u = ((uint32_t)h) << 16;
    return c.f;
}
__device__ __forceinline__ v8f v8f_zero() {
    v8f z = {0.f,0.f,0.f,0.f,0.f,0.f,0.f,0.f};
    return z;
}

// ---------------- f32 -> bf16 convert ------------
__global__ __launch_bounds__(256) void cvt_kernel(const float* __restrict__ in,
                                                  uint16_t* __restrict__ out, int n) {
    int i = blockIdx.x * 256 + threadIdx.x;
    if (i < n) out[i] = f2bf(in[i]);
}

// ---------------- positional encoding ------------
__global__ __launch_bounds__(256) void posenc_kernel(const float* __restrict__ src,
                                                     float* __restrict__ xf,
                                                     uint16_t* __restrict__ xb) {
    int idx = blockIdx.x * 256 + threadIdx.x;          // over B*L*(D/2)
    if (idx >= BB * LL * (DD / 2)) return;
    int i = idx % (DD / 2);
    int l = (idx / (DD / 2)) % LL;
    float div = __expf((2.0f * i) * (-logf(10000.0f) / (float)DD));
    float ang = (float)l * div;
    size_t base = (size_t)(idx / (DD / 2)) * DD + 2 * i;
    float v0 = src[base]     + __sinf(ang);
    float v1 = src[base + 1] + __cosf(ang);
    xf[base] = v0; xf[base + 1] = v1;
    xb[base] = f2bf(v0); xb[base + 1] = f2bf(v1);
}

// ---------------- bf16 WMMA GEMM -----------------
// C[M,N] = A[M,K](bf16,row-major) * W[N,K]^T (bf16,row-major) (+bias)(+add)(act)
// block: 256 threads = 8 waves; wave -> 16x64 tile; grid = (M/128, ceil(N/64))
// Branch-free K loop: tail N-tiles clamp their B row pointer to row 0 and the
// (garbage) results are simply not stored. act: 0 none, 1 relu, 2 softplus.
__global__ __launch_bounds__(256) void gemm_bf16_kernel(
    const uint16_t* __restrict__ A, int lda,
    const uint16_t* __restrict__ W, int ldw,
    float* __restrict__ Cf, uint16_t* __restrict__ Cb, int ldc,
    const float* __restrict__ bias, const float* __restrict__ add,
    int N, int K, int act)
{
    int lane = threadIdx.x & 31;
    int wave = threadIdx.x >> 5;
    int half = lane >> 4;
    int rm   = lane & 15;
    int row0 = blockIdx.x * 128 + wave * 16;
    int n0   = blockIdx.y * 64;

    bool tv[4];
    const uint16_t* wrow[4];
#pragma unroll
    for (int t = 0; t < 4; ++t) {
        int nb = n0 + t * 16;
        tv[t] = nb < N;
        if (!tv[t]) nb = 0;                       // clamp: loads stay in-bounds
        wrow[t] = W + (size_t)(nb + rm) * ldw + 8 * half;
    }

    v8f acc0 = v8f_zero(), acc1 = v8f_zero(), acc2 = v8f_zero(), acc3 = v8f_zero();

    const uint16_t* arow = A + (size_t)(row0 + rm) * lda + 8 * half;

    for (int kk = 0; kk < K; kk += 32) {
        AFrag a;
        a.h[0] = *(const u16x8*)(arow + kk);
        a.h[1] = *(const u16x8*)(arow + kk + 16);
        __builtin_prefetch(arow + kk + 32, 0, 3); // speculative; dropped if OOB

        AFrag b0, b1, b2, b3;
        b0.h[0] = *(const u16x8*)(wrow[0] + kk);
        b0.h[1] = *(const u16x8*)(wrow[0] + kk + 16);
        b1.h[0] = *(const u16x8*)(wrow[1] + kk);
        b1.h[1] = *(const u16x8*)(wrow[1] + kk + 16);
        b2.h[0] = *(const u16x8*)(wrow[2] + kk);
        b2.h[1] = *(const u16x8*)(wrow[2] + kk + 16);
        b3.h[0] = *(const u16x8*)(wrow[3] + kk);
        b3.h[1] = *(const u16x8*)(wrow[3] + kk + 16);

        acc0 = __builtin_amdgcn_wmma_f32_16x16x32_bf16(false, a.bf, false, b0.bf, (short)0, acc0, false, false);
        acc1 = __builtin_amdgcn_wmma_f32_16x16x32_bf16(false, a.bf, false, b1.bf, (short)0, acc1, false, false);
        acc2 = __builtin_amdgcn_wmma_f32_16x16x32_bf16(false, a.bf, false, b2.bf, (short)0, acc2, false, false);
        acc3 = __builtin_amdgcn_wmma_f32_16x16x32_bf16(false, a.bf, false, b3.bf, (short)0, acc3, false, false);
    }

    v8f accs[4] = { acc0, acc1, acc2, acc3 };
#pragma unroll
    for (int t = 0; t < 4; ++t) {
        if (!tv[t]) continue;                      // uniform, epilogue only
        int n = n0 + t * 16 + rm;
        float bv = bias ? bias[n] : 0.0f;
#pragma unroll
        for (int v = 0; v < 8; ++v) {
            int m = row0 + v + 8 * half;
            size_t idx = (size_t)m * ldc + n;
            float val = accs[t][v] + bv;
            if (add) val += add[idx];
            if (act == 1)      val = fmaxf(val, 0.0f);
            else if (act == 2) val = (val > 20.0f) ? val : log1pf(__expf(val));
            if (Cf) Cf[idx] = val;
            if (Cb) Cb[idx] = f2bf(val);
        }
    }
}

// ---------------- flash attention ----------------
// 1 wave per (q-tile 16, head, batch). QKV bf16 (B,L,3D); O bf16 (B,L,D)
__global__ __launch_bounds__(32) void attn_kernel(const uint16_t* __restrict__ QKV,
                                                  uint16_t* __restrict__ O)
{
    const int rs = 3 * DD;                 // row stride of QKV
    int lane = threadIdx.x;
    int half = lane >> 4;
    int rm   = lane & 15;
    int q0 = blockIdx.x * 16;
    int h  = blockIdx.y;
    int b  = blockIdx.z;

    const uint16_t* qb = QKV + (size_t)b * LL * rs + h * HD;
    const uint16_t* kb = qb + DD;
    const uint16_t* vb = qb + 2 * DD;

    AFrag aq[2];
#pragma unroll
    for (int c = 0; c < 2; ++c) {
        const uint16_t* p = qb + (size_t)(q0 + rm) * rs + 32 * c + 8 * half;
        aq[c].h[0] = *(const u16x8*)p;
        aq[c].h[1] = *(const u16x8*)(p + 16);
    }

    float mrow[8], lrow[8];
    v8f oacc[4];
#pragma unroll
    for (int v = 0; v < 8; ++v) { mrow[v] = -1e30f; lrow[v] = 0.0f; }
#pragma unroll
    for (int t = 0; t < 4; ++t) oacc[t] = v8f_zero();

    __shared__ uint16_t pl[16][32];   // P tile, row-major (m, kv)
    __shared__ uint16_t vT[64][32];   // V chunk transposed (d, kv)

    const float scale = 0.125f;       // 1/sqrt(64)

    for (int j = 0; j < LL / 32; ++j) {
        int kv0 = j * 32;
        v8f s[2];
#pragma unroll
        for (int sc = 0; sc < 2; ++sc) {
            const uint16_t* kp = kb + (size_t)(kv0 + sc * 16 + rm) * rs + 8 * half;
            AFrag b0, b1;
            b0.h[0] = *(const u16x8*)kp;        b0.h[1] = *(const u16x8*)(kp + 16);
            b1.h[0] = *(const u16x8*)(kp + 32); b1.h[1] = *(const u16x8*)(kp + 48);
            v8f z = v8f_zero();
            z = __builtin_amdgcn_wmma_f32_16x16x32_bf16(false, aq[0].bf, false, b0.bf, (short)0, z, false, false);
            z = __builtin_amdgcn_wmma_f32_16x16x32_bf16(false, aq[1].bf, false, b1.bf, (short)0, z, false, false);
            s[sc] = z;
        }
#pragma unroll
        for (int v = 0; v < 8; ++v) {
            float a0 = s[0][v] * scale;
            float a1 = s[1][v] * scale;
            float mx = fmaxf(a0, a1);
#pragma unroll
            for (int off = 1; off < 16; off <<= 1) mx = fmaxf(mx, __shfl_xor(mx, off));
            float nm   = fmaxf(mrow[v], mx);
            float corr = __expf(mrow[v] - nm);
            float p0   = __expf(a0 - nm);
            float p1   = __expf(a1 - nm);
            float ps   = p0 + p1;
#pragma unroll
            for (int off = 1; off < 16; off <<= 1) ps += __shfl_xor(ps, off);
            lrow[v] = lrow[v] * corr + ps;
            mrow[v] = nm;
#pragma unroll
            for (int t = 0; t < 4; ++t) oacc[t][v] *= corr;
            int m = v + 8 * half;
            pl[m][rm]      = f2bf(p0);
            pl[m][16 + rm] = f2bf(p1);
        }
        {
            const uint16_t* vp = vb + (size_t)(kv0 + lane) * rs;
#pragma unroll
            for (int dd = 0; dd < 64; dd += 8) {
                u16x8 tv8 = *(const u16x8*)(vp + dd);
#pragma unroll
                for (int e = 0; e < 8; ++e) vT[dd + e][lane] = tv8[e];
            }
        }
        __syncthreads();
        AFrag ap;
        {
            const uint16_t* pp = &pl[rm][8 * half];
            ap.h[0] = *(const u16x8*)pp;
            ap.h[1] = *(const u16x8*)(pp + 16);
        }
#pragma unroll
        for (int t = 0; t < 4; ++t) {
            const uint16_t* vp = &vT[t * 16 + rm][8 * half];
            AFrag bv;
            bv.h[0] = *(const u16x8*)vp;
            bv.h[1] = *(const u16x8*)(vp + 16);
            oacc[t] = __builtin_amdgcn_wmma_f32_16x16x32_bf16(
                false, ap.bf, false, bv.bf, (short)0, oacc[t], false, false);
        }
        __syncthreads();
    }

#pragma unroll
    for (int t = 0; t < 4; ++t)
#pragma unroll
        for (int v = 0; v < 8; ++v) {
            int m = v + 8 * half;
            float val = oacc[t][v] / lrow[v];
            O[((size_t)b * LL + q0 + m) * DD + h * HD + t * 16 + rm] = f2bf(val);
        }
}

// ---------------- depthwise causal conv + SiLU ----
__global__ __launch_bounds__(256) void conv_kernel(const uint16_t* __restrict__ XZ,
                                                   const float* __restrict__ cw,
                                                   const float* __restrict__ cb,
                                                   float* __restrict__ XCf,
                                                   uint16_t* __restrict__ XCb)
{
    int idx = blockIdx.x * 256 + threadIdx.x;          // over B*L*DI
    if (idx >= BB * LL * DI) return;
    int di = idx % DI;
    int l  = (idx / DI) % LL;
    int b  = idx / (DI * LL);
    float s = cb[di];
#pragma unroll
    for (int j = 0; j < DC; ++j) {
        int ls = l - (DC - 1) + j;
        if (ls >= 0)
            s += bf2f(XZ[((size_t)b * LL + ls) * (2 * DI) + di]) * cw[di * DC + j];
    }
    float sil = s / (1.0f + __expf(-s));
    XCf[idx] = sil;
    XCb[idx] = f2bf(sil);
}

// ---------------- selective scan ------------------
__global__ __launch_bounds__(256) void scan_kernel(const float* __restrict__ XDBL,
                                                   const float* __restrict__ DT,
                                                   const float* __restrict__ XC,
                                                   const float* __restrict__ Alog,
                                                   float* __restrict__ YS)
{
    int tid = threadIdx.x;
    int g   = tid >> 2;
    int sub = tid & 3;
    int di  = blockIdx.x * 64 + g;
    int b   = blockIdx.y;

    float Ac[16], hs[16];
#pragma unroll
    for (int jj = 0; jj < 16; ++jj) {
        Ac[jj] = -__expf(Alog[(size_t)di * DS + sub * 16 + jj]);
        hs[jj] = 0.0f;
    }
    __shared__ float sB[64], sC[64];

    for (int l = 0; l < LL; ++l) {
        size_t r160 = ((size_t)b * LL + l) * (DTR + 2 * DS);
        if (tid < 64)       sB[tid]      = XDBL[r160 + DTR + tid];
        else if (tid < 128) sC[tid - 64] = XDBL[r160 + DTR + DS + (tid - 64)];
        __syncthreads();
        size_t rdi = ((size_t)b * LL + l) * DI + di;
        float dt = DT[rdi];
        float u  = XC[rdi];
        float du = dt * u;
        float acc = 0.0f;
#pragma unroll
        for (int jj = 0; jj < 16; ++jj) {
            float e = __expf(dt * Ac[jj]);
            hs[jj] = e * hs[jj] + du * sB[sub * 16 + jj];
            acc += hs[jj] * sC[sub * 16 + jj];
        }
        acc += __shfl_xor(acc, 1);
        acc += __shfl_xor(acc, 2);
        if (sub == 0) YS[rdi] = acc;
        __syncthreads();
    }
}

// ---------------- mamba gating --------------------
__global__ __launch_bounds__(256) void gate_kernel(const float* __restrict__ YS,
                                                   const float* __restrict__ XC,
                                                   const float* __restrict__ Dsk,
                                                   const uint16_t* __restrict__ XZ,
                                                   uint16_t* __restrict__ Mb)
{
    int idx = blockIdx.x * 256 + threadIdx.x;          // over B*L*DI
    if (idx >= BB * LL * DI) return;
    int di  = idx % DI;
    int row = idx / DI;
    float y = YS[idx] + XC[idx] * Dsk[di];
    float z = bf2f(XZ[(size_t)row * (2 * DI) + DI + di]);
    float sz = z / (1.0f + __expf(-z));
    Mb[idx] = f2bf(y * sz);
}

// ---------------- layer norm ----------------------
__global__ __launch_bounds__(256) void ln_kernel(const float* __restrict__ in,
                                                 const float* __restrict__ g,
                                                 const float* __restrict__ bb,
                                                 float* __restrict__ outf,
                                                 uint16_t* __restrict__ outb)
{
    int row = blockIdx.x;
    int tid = threadIdx.x;
    const float* r = in + (size_t)row * DD;
    float x0 = r[tid], x1 = r[tid + 256];
    float s  = x0 + x1;
    float sq = x0 * x0 + x1 * x1;
#pragma unroll
    for (int off = 1; off < 32; off <<= 1) {
        s  += __shfl_xor(s,  off);
        sq += __shfl_xor(sq, off);
    }
    __shared__ float ws[8], wq[8];
    int w = tid >> 5;
    if ((tid & 31) == 0) { ws[w] = s; wq[w] = sq; }
    __syncthreads();
    if (tid == 0) {
        float S = 0.f, Q = 0.f;
        for (int i = 0; i < 8; ++i) { S += ws[i]; Q += wq[i]; }
        float mean = S / (float)DD;
        float var  = Q / (float)DD - mean * mean;
        ws[0] = mean;
        wq[0] = 1.0f / sqrtf(var + 1e-5f);
    }
    __syncthreads();
    float mean = ws[0], rstd = wq[0];
    float y0 = (x0 - mean) * rstd * g[tid]       + bb[tid];
    float y1 = (x1 - mean) * rstd * g[tid + 256] + bb[tid + 256];
    size_t base = (size_t)row * DD;
    if (outf) { outf[base + tid] = y0; outf[base + tid + 256] = y1; }
    if (outb) { outb[base + tid] = f2bf(y0); outb[base + tid + 256] = f2bf(y1); }
}

// ================= host launch ====================
extern "C" void kernel_launch(void* const* d_in, const int* in_sizes, int n_in,
                              void* d_out, int out_size, void* d_ws, size_t ws_size,
                              hipStream_t stream)
{
    (void)in_sizes; (void)n_in; (void)out_size; (void)ws_size;

    const float* src      = (const float*)d_in[0];
    const float* w_qkv    = (const float*)d_in[1];
    const float* b_qkv    = (const float*)d_in[2];
    const float* w_o      = (const float*)d_in[3];
    const float* b_o      = (const float*)d_in[4];
    const float* w1       = (const float*)d_in[5];
    const float* b1       = (const float*)d_in[6];
    const float* w2       = (const float*)d_in[7];
    const float* b2       = (const float*)d_in[8];
    const float* n1g      = (const float*)d_in[9];
    const float* n1b      = (const float*)d_in[10];
    const float* w_in     = (const float*)d_in[11];
    const float* conv_w   = (const float*)d_in[12];
    const float* conv_b   = (const float*)d_in[13];
    const float* w_xp     = (const float*)d_in[14];
    const float* w_dt     = (const float*)d_in[15];
    const float* b_dt     = (const float*)d_in[16];
    const float* A_log    = (const float*)d_in[17];
    const float* D_skip   = (const float*)d_in[18];
    const float* w_op     = (const float*)d_in[19];
    const float* nfg      = (const float*)d_in[20];
    const float* nfb      = (const float*)d_in[21];

    char* wsb = (char*)d_ws;
    size_t off = 0;
    auto alloc = [&](size_t bytes) -> void* {
        off = (off + 255) & ~(size_t)255;
        void* p = wsb + off;
        off += bytes;
        return p;
    };
    float*    Xf    = (float*)   alloc((size_t)ROWS * DD * 4);
    uint16_t* Xb    = (uint16_t*)alloc((size_t)ROWS * DD * 2);
    uint16_t* QKVb  = (uint16_t*)alloc((size_t)ROWS * 3 * DD * 2);
    uint16_t* Ob    = (uint16_t*)alloc((size_t)ROWS * DD * 2);
    float*    Af    = (float*)   alloc((size_t)ROWS * DD * 4);
    uint16_t* XZb   = (uint16_t*)alloc((size_t)ROWS * 2 * DI * 2);
    float*    XCf   = (float*)   alloc((size_t)ROWS * DI * 4);
    uint16_t* XCb   = (uint16_t*)alloc((size_t)ROWS * DI * 2);
    float*    XDBLf = (float*)   alloc((size_t)ROWS * 160 * 4);
    uint16_t* XDBLb = (uint16_t*)alloc((size_t)ROWS * 160 * 2);
    float*    DTf   = (float*)   alloc((size_t)ROWS * DI * 4);
    float*    YSf   = (float*)   alloc((size_t)ROWS * DI * 4);
    uint16_t* Mb    = (uint16_t*)alloc((size_t)ROWS * DI * 2);
    float*    Hf    = (float*)   alloc((size_t)ROWS * DD * 4);
    uint16_t* Hb    = (uint16_t*)alloc((size_t)ROWS * DD * 2);
    uint16_t* U1b   = (uint16_t*)alloc((size_t)ROWS * FF * 2);
    float*    H2f   = (float*)   alloc((size_t)ROWS * DD * 4);
    uint16_t* Wqkv = (uint16_t*)alloc((size_t)NL * 3 * DD * DD * 2);
    uint16_t* Wwo  = (uint16_t*)alloc((size_t)NL * DD * DD * 2);
    uint16_t* Ww1  = (uint16_t*)alloc((size_t)NL * FF * DD * 2);
    uint16_t* Ww2  = (uint16_t*)alloc((size_t)NL * DD * FF * 2);
    uint16_t* Win  = (uint16_t*)alloc((size_t)NL * 2 * DI * DD * 2);
    uint16_t* Wxp  = (uint16_t*)alloc((size_t)NL * 160 * DI * 2);
    uint16_t* Wdt  = (uint16_t*)alloc((size_t)NL * DI * DTR * 2);
    uint16_t* Wop  = (uint16_t*)alloc((size_t)NL * DD * DI * 2);

    auto cvt = [&](const float* in, uint16_t* out, size_t n) {
        cvt_kernel<<<(int)((n + 255) / 256), 256, 0, stream>>>(in, out, (int)n);
    };
    cvt(w_qkv, Wqkv, (size_t)NL * 3 * DD * DD);
    cvt(w_o,   Wwo,  (size_t)NL * DD * DD);
    cvt(w1,    Ww1,  (size_t)NL * FF * DD);
    cvt(w2,    Ww2,  (size_t)NL * DD * FF);
    cvt(w_in,  Win,  (size_t)NL * 2 * DI * DD);
    cvt(w_xp,  Wxp,  (size_t)NL * 160 * DI);
    cvt(w_dt,  Wdt,  (size_t)NL * DI * DTR);
    cvt(w_op,  Wop,  (size_t)NL * DD * DI);

    posenc_kernel<<<(BB * LL * (DD / 2) + 255) / 256, 256, 0, stream>>>(src, Xf, Xb);

    auto gemm = [&](const uint16_t* A, int lda, const uint16_t* W, int ldw,
                    float* Cf, uint16_t* Cb, int ldc,
                    const float* bias, const float* add, int N, int K, int act) {
        dim3 grid(ROWS / 128, (N + 63) / 64);
        gemm_bf16_kernel<<<grid, 256, 0, stream>>>(A, lda, W, ldw, Cf, Cb, ldc,
                                                   bias, add, N, K, act);
    };

    for (int l = 0; l < NL; ++l) {
        // ---- attention branch ----
        gemm(Xb, DD, Wqkv + (size_t)l * 3 * DD * DD, DD,
             nullptr, QKVb, 3 * DD, b_qkv + (size_t)l * 3 * DD, nullptr, 3 * DD, DD, 0);
        attn_kernel<<<dim3(LL / 16, NH, BB), 32, 0, stream>>>(QKVb, Ob);
        gemm(Ob, DD, Wwo + (size_t)l * DD * DD, DD,
             Af, nullptr, DD, b_o + (size_t)l * DD, nullptr, DD, DD, 0);

        // ---- mamba branch ----
        gemm(Xb, DD, Win + (size_t)l * 2 * DI * DD, DD,
             nullptr, XZb, 2 * DI, nullptr, nullptr, 2 * DI, DD, 0);
        conv_kernel<<<(BB * LL * DI + 255) / 256, 256, 0, stream>>>(
            XZb, conv_w + (size_t)l * DI * DC, conv_b + (size_t)l * DI, XCf, XCb);
        gemm(XCb, DI, Wxp + (size_t)l * 160 * DI, DI,
             XDBLf, XDBLb, 160, nullptr, nullptr, 160, DI, 0);
        gemm(XDBLb, 160, Wdt + (size_t)l * DI * DTR, DTR,
             DTf, nullptr, DI, b_dt + (size_t)l * DI, nullptr, DI, DTR, 2);
        scan_kernel<<<dim3(DI / 64, BB), 256, 0, stream>>>(
            XDBLf, DTf, XCf, A_log + (size_t)l * DI * DS, YSf);
        gate_kernel<<<(BB * LL * DI + 255) / 256, 256, 0, stream>>>(
            YSf, XCf, D_skip + (size_t)l * DI, XZb, Mb);
        gemm(Mb, DI, Wop + (size_t)l * DD * DI, DI,
             Hf, Hb, DD, nullptr, Af, DD, DI, 0);

        // ---- FFN + norm ----
        gemm(Hb, DD, Ww1 + (size_t)l * FF * DD, DD,
             nullptr, U1b, FF, b1 + (size_t)l * FF, nullptr, FF, DD, 1);
        gemm(U1b, FF, Ww2 + (size_t)l * DD * FF, FF,
             H2f, nullptr, DD, b2 + (size_t)l * DD, nullptr, DD, FF, 0);
        ln_kernel<<<ROWS, 256, 0, stream>>>(
            H2f, n1g + (size_t)l * DD, n1b + (size_t)l * DD, Xf, Xb);
    }

    ln_kernel<<<ROWS, 256, 0, stream>>>(Xf, nfg, nfb, (float*)d_out, nullptr);
}
